// BioNormalizedPolynomialCKN1D_54288386621701
// MI455X (gfx1250) — matrix operand order repaired and checked by reading, not compiled, not run on hardware
//
#include <hip/hip_runtime.h>
#include <stdint.h>

// ---------------------------------------------------------------------------
// BioNormalizedPolynomialCKN1D on MI455X (gfx1250)
//   out[b,t,f] = (g * dot[b,t,f] / sqrt(ss[b,t] + eps) + c)^2 + bias[f]
//   dot = 1D VALID conv (K=7, Cin=64, F=128) -> implicit GEMM, Kdim=448
//   ss  = windowed sum of squares, filter-independent (computed once per t)
// GEMM on v_wmma_f32_16x16x32_bf16 (f32 accumulate); norm/epilogue in fp32.
// v2: two 16-position t-tiles per iteration (28 wmma / sync round),
//     per-t reciprocal norm (multiplies instead of per-lane divides).
// ---------------------------------------------------------------------------

typedef __bf16 bf16_t;
typedef bf16_t v16bf __attribute__((ext_vector_type(16)));
typedef float  v8f   __attribute__((ext_vector_type(8)));

#define KSIZE 7
#define CIN   64
#define FILT  128
#define LIN   4096
#define LOUT  4090
#define KTOT  (KSIZE * CIN)        // 448
#define NCHUNK (KTOT / 32)         // 14 WMMA K-steps
#define FH    64                   // filters handled per workgroup
#define TILE_T 16
#define IT_T  32                   // 2 t-tiles per iteration
#define STRIP_IT 8                 // 8 iters * 32 = 256 positions / workgroup
#define NROWS (IT_T + KSIZE - 1)   // 38 staged x rows per iteration
#define EPSV  1e-7f

__device__ __forceinline__ uint32_t pack_bf16(float a, float b) {
  // round-to-nearest-even f32 -> bf16, packed low/high
  uint32_t ua = __float_as_uint(a);
  uint32_t ub = __float_as_uint(b);
  uint32_t ra = (ua + 0x7FFFu + ((ua >> 16) & 1u)) >> 16;
  uint32_t rb = (ub + 0x7FFFu + ((ub >> 16) & 1u)) >> 16;
  return (ra & 0xFFFFu) | ((rb & 0xFFFFu) << 16);
}

union BFrag { uint32_t u[8]; v16bf v; };

__global__ __launch_bounds__(128, 1)
void ckn1d_wmma_kernel(const float* __restrict__ xg,   // [64,4096,64]
                       const float* __restrict__ kg,   // [7,64,128]
                       const float* __restrict__ bg,   // [128]
                       const float* __restrict__ gp,   // scalar
                       const float* __restrict__ cp,   // scalar
                       float* __restrict__ outg)       // [64,4090,128]
{
  // K pair-packed weights: lds_w[p*FH + f] = {w[2p][f], w[2p+1][f]} as bf16x2
  __shared__ uint32_t lds_w[(KTOT / 2) * FH];          // 57344 B
  __shared__ uint32_t lds_x[NROWS * (CIN / 2)];        //  4864 B
  __shared__ float    row_ss[NROWS];
  __shared__ float    t_inv[IT_T];

  const int tid   = threadIdx.x;
  const int lane  = tid & 31;
  const int wave  = tid >> 5;       // 4 waves: filter sub-tile
  const int strip = blockIdx.x;     // 16 strips of 256 t-positions
  const int fh    = blockIdx.y;     // 2 filter halves of 64
  const int batch = blockIdx.z;     // 64 batches

  const float gmul = gp[0];
  const float cadd = cp[0];

  // ---- stage this block's 64 filters into LDS as bf16 K-pairs -------------
  for (int idx = tid; idx < (KTOT / 2) * FH; idx += 128) {
    int p  = idx >> 6;              // K pair index (0..223)
    int f  = idx & 63;              // local filter
    int kk = p * 2;                 // flattened (kw, ci)
    int kw = kk / CIN;
    int ci = kk % CIN;
    size_t gidx = (size_t)kw * (CIN * FILT) + (size_t)ci * FILT + (fh * FH + f);
    lds_w[idx] = pack_bf16(kg[gidx], kg[gidx + FILT]); // ci, ci+1 same kw
  }
  __syncthreads();

  // ---- hoist B fragments into registers: reused across the whole strip ---
  // B (32x16, bf16): lanes 0-15 hold K=0..15, lanes 16-31 hold K=16..31,
  // column N = lane%16, VGPR j holds K pair (2j, 2j+1) within the half.
  const int lgrp = lane >> 4;
  const int lm   = lane & 15;
  BFrag bfr[NCHUNK];
#pragma unroll
  for (int kc = 0; kc < NCHUNK; ++kc) {
#pragma unroll
    for (int j = 0; j < 8; ++j) {
      int p = kc * 16 + lgrp * 8 + j;                 // global K/2
      bfr[kc].u[j] = lds_w[p * FH + wave * 16 + lm];
    }
  }

  const float  fbias = bg[fh * FH + wave * 16 + lm];
  const size_t xbase = (size_t)batch * LIN * CIN;

  for (int it = 0; it < STRIP_IT; ++it) {
    const int t0 = strip * (STRIP_IT * IT_T) + it * IT_T;

    __syncthreads();                          // previous iteration consumed
    if (tid < NROWS) row_ss[tid] = 0.0f;
    __syncthreads();

    // ---- stage x rows t0..t0+37 (clamped) + per-row sum of squares -------
    for (int idx = tid; idx < NROWS * (CIN / 2); idx += 128) {
      int r   = idx >> 5;                     // local row 0..37
      int cpi = idx & 31;                     // channel pair 0..31
      int row = t0 + r; if (row > LIN - 1) row = LIN - 1;
      const float* src = xg + xbase + (size_t)row * CIN + cpi * 2;
      float f0 = src[0];
      float f1 = src[1];
      lds_x[idx] = pack_bf16(f0, f1);
      atomicAdd(&row_ss[r], f0 * f0 + f1 * f1);
      __builtin_prefetch(src + IT_T * CIN, 0, 1);     // next iteration's rows
    }
    __syncthreads();

    // per-t reciprocal norm: one precise sqrt+div per t, shared by 64 filters
    if (tid < IT_T) {
      float s = EPSV;
#pragma unroll
      for (int kw = 0; kw < KSIZE; ++kw) s += row_ss[tid + kw];
      t_inv[tid] = 1.0f / sqrtf(s);
    }
    __syncthreads();

    // ---- 28 x v_wmma_f32_16x16x32_bf16 over the 448-wide K dimension -----
    v8f acc0 = {};
    v8f acc1 = {};
#pragma unroll
    for (int kc = 0; kc < NCHUNK; ++kc) {
      const int kw = kc >> 1;                 // chunk lies inside one tap
      const int cb = (kc & 1) * 16;           // channel-pair base (0 or 16)
      BFrag a0, a1;
#pragma unroll
      for (int j = 0; j < 8; ++j) {
        // A 16x32 bf16 layout: VGPR j holds K pair; pair offset within chunk:
        //   o/2 = (lane/16)*8 + j
        int o2 = lgrp * 8 + j;
        int col = cb + o2;
        a0.u[j] = lds_x[(lm + kw) * (CIN / 2) + col];
        a1.u[j] = lds_x[(lm + kw + TILE_T) * (CIN / 2) + col];
      }
      acc0 = __builtin_amdgcn_wmma_f32_16x16x32_bf16(
               false, a0.v, false, bfr[kc].v, (short)0, acc0, false, false);
      acc1 = __builtin_amdgcn_wmma_f32_16x16x32_bf16(
               false, a1.v, false, bfr[kc].v, (short)0, acc1, false, false);
    }

    // ---- epilogue: C/D VGPR i -> M = (lane/16)*8 + i, N = lane%16 --------
    const int fglob = fh * FH + wave * 16 + lm;
#pragma unroll
    for (int i = 0; i < 8; ++i) {
      int m = lgrp * 8 + i;
      int t = t0 + m;
      if (t < LOUT) {
        float q = gmul * (acc0[i] * t_inv[m]) + cadd;
        outg[((size_t)batch * LOUT + t) * FILT + fglob] = q * q + fbias;
      }
    }
#pragma unroll
    for (int i = 0; i < 8; ++i) {
      int m = TILE_T + lgrp * 8 + i;
      int t = t0 + m;
      if (t < LOUT) {
        float q = gmul * (acc1[i] * t_inv[m]) + cadd;
        outg[((size_t)batch * LOUT + t) * FILT + fglob] = q * q + fbias;
      }
    }
  }
}

extern "C" void kernel_launch(void* const* d_in, const int* in_sizes, int n_in,
                              void* d_out, int out_size, void* d_ws, size_t ws_size,
                              hipStream_t stream) {
  (void)in_sizes; (void)n_in; (void)out_size; (void)d_ws; (void)ws_size;
  const float* x = (const float*)d_in[0];
  const float* k = (const float*)d_in[1];
  const float* b = (const float*)d_in[2];
  const float* g = (const float*)d_in[3];
  const float* c = (const float*)d_in[4];
  float* out = (float*)d_out;

  dim3 grid(16 /*t strips*/, 2 /*filter halves*/, 64 /*batch*/);
  dim3 block(128);
  ckn1d_wmma_kernel<<<grid, block, 0, stream>>>(x, k, b, g, c, out);
}
